// TransformerBlock_81836306858722
// MI455X (gfx1250) — compile-verified, matrix operand order checked by reading, not run on hardware
//
#include <hip/hip_runtime.h>
#include <hip/hip_bf16.h>
#include <math.h>

// ---------------- problem constants ----------------
constexpr int CB  = 2;
constexpr int CS  = 2048;
constexpr int CD  = 1024;
constexpr int CNH = 16;
constexpr int CHD = 64;
constexpr int CIN = 4096;
constexpr int CM  = CB * CS;          // 4096 token rows

// ---------------- types ----------------
typedef __attribute__((ext_vector_type(16))) __bf16     bf16x16;
typedef __attribute__((ext_vector_type(8)))  float      f32x8;
typedef __attribute__((ext_vector_type(4)))  unsigned   u32x4;

__device__ __forceinline__ __bf16 f2bf(float f) {
  union { float f; unsigned u; } v; v.f = f;
  unsigned r = (v.u + 0x7FFFu + ((v.u >> 16) & 1u)) >> 16;
  unsigned short s = (unsigned short)r;
  __bf16 b; __builtin_memcpy(&b, &s, 2); return b;
}
__device__ __forceinline__ float bf2f(__bf16 b) {
  unsigned short s; __builtin_memcpy(&s, &b, 2);
  union { unsigned u; float f; } v; v.u = ((unsigned)s) << 16; return v.f;
}

// 16x32 bf16 fragment load (A or B operand), uniform base + 32-bit lane offset
// so the backend can select the saddr+voffset global-load form.
// Wave32 layout (ISA 7.12.2): lane holds row (lane&15); half-wave selects the
// two contiguous 8-element K chunks at half*8 and 16+half*8.
__device__ __forceinline__ bf16x16 load_frag_u(const __bf16* __restrict__ base,
                                               unsigned off, int half) {
  bf16x16 f;
  u32x4* fp = reinterpret_cast<u32x4*>(&f);
  const __bf16* p = base + off + half * 8;
  fp[0] = *reinterpret_cast<const u32x4*>(p);
  fp[1] = *reinterpret_cast<const u32x4*>(p + 16);
  return f;
}

__device__ __forceinline__ f32x8 wmma_bf16(bf16x16 a, bf16x16 b, f32x8 c) {
  return __builtin_amdgcn_wmma_f32_16x16x32_bf16(false, a, false, b, (short)0, c,
                                                 false, false);
}
__device__ __forceinline__ f32x8 zero8() {
  f32x8 z;
#pragma unroll
  for (int j = 0; j < 8; ++j) z[j] = 0.f;
  return z;
}

// -------- weight convert + transpose, 32x32 LDS tile (fp32 RxC -> bf16 CxR) --
// R, C are multiples of 32.  Both global accesses fully coalesced.
__global__ __launch_bounds__(256) void conv_transpose(const float* __restrict__ s,
                                                      __bf16* __restrict__ d,
                                                      int R, int C) {
  __shared__ __bf16 tile[32][33];
  const int tx = threadIdx.x & 31, ty = threadIdx.x >> 5;  // 32 x 8
  const int c0 = blockIdx.x * 32, r0 = blockIdx.y * 32;
#pragma unroll
  for (int i = 0; i < 32; i += 8)
    tile[ty + i][tx] = f2bf(s[(size_t)(r0 + ty + i) * C + (c0 + tx)]);
  __syncthreads();
#pragma unroll
  for (int i = 0; i < 32; i += 8)
    d[(size_t)(c0 + ty + i) * R + (r0 + tx)] = tile[tx][ty + i];
}

// ---------------- RMSNorm: fp32 row -> bf16 row ----------------
__global__ __launch_bounds__(256) void rmsnorm_bf16(const float* __restrict__ x,
                                                    const float* __restrict__ w,
                                                    __bf16* __restrict__ o) {
  int row = blockIdx.x;
  const float* xr = x + (size_t)row * CD;
  float acc = 0.f;
  for (int c = threadIdx.x; c < CD; c += 256) { float v = xr[c]; acc += v * v; }
#pragma unroll
  for (int m = 16; m >= 1; m >>= 1) acc += __shfl_xor(acc, m, 32);
  __shared__ float red[8];
  if ((threadIdx.x & 31) == 0) red[threadIdx.x >> 5] = acc;
  __syncthreads();
  float tot = 0.f;
#pragma unroll
  for (int i = 0; i < 8; ++i) tot += red[i];
  float rms = rsqrtf(tot / (float)CD + 1e-6f);
  for (int c = threadIdx.x; c < CD; c += 256)
    o[(size_t)row * CD + c] = f2bf(xr[c] * rms * w[c]);
}

// ---------------- generic WMMA GEMM:  C[MxN] = A[MxK] * Bt[NxK]^T ------------
// Software-pipelined (ping-pong fragment buffers) so next-chunk loads fill the
// WMMA->VALU WAR hazard slots instead of v_nops.  K % 64 == 0.
// MODE 0: outBF = bf16(C).  MODE 1: outF = resid + C (fp32).
template <int MODE>
__global__ __launch_bounds__(256) void gemm_wmma(
    const __bf16* __restrict__ A, const __bf16* __restrict__ Bt,
    int M, int N, int K,
    const float* __restrict__ resid, float* __restrict__ outF,
    __bf16* __restrict__ outBF) {
  const int lane = threadIdx.x & 31;
  const int half = lane >> 4;
  const int wave = threadIdx.x >> 5;                 // 8 waves: 4 (M) x 2 (N)
  const int m0 = blockIdx.y * 128 + (wave & 3) * 32;
  const int n0 = blockIdx.x * 128 + (wave >> 2) * 64;

  f32x8 c[2][4];
#pragma unroll
  for (int mi = 0; mi < 2; ++mi)
#pragma unroll
    for (int ni = 0; ni < 4; ++ni) c[mi][ni] = zero8();

  unsigned aoff[2], boff[4];
  aoff[0] = (unsigned)(m0 + (lane & 15)) * (unsigned)K;
  aoff[1] = aoff[0] + (unsigned)(16 * K);
#pragma unroll
  for (int ni = 0; ni < 4; ++ni)
    boff[ni] = (unsigned)(n0 + ni * 16 + (lane & 15)) * (unsigned)K;

  bf16x16 aB[2][2], bB[2][4];
#pragma unroll
  for (int mi = 0; mi < 2; ++mi) aB[0][mi] = load_frag_u(A, aoff[mi], half);
#pragma unroll
  for (int ni = 0; ni < 4; ++ni) bB[0][ni] = load_frag_u(Bt, boff[ni], half);

  for (int k0 = 0; k0 < K; k0 += 64) {
    // prefetch buffer 1 @ k0+32 (always in range: K % 64 == 0)
#pragma unroll
    for (int mi = 0; mi < 2; ++mi)
      aB[1][mi] = load_frag_u(A, aoff[mi] + k0 + 32, half);
#pragma unroll
    for (int ni = 0; ni < 4; ++ni)
      bB[1][ni] = load_frag_u(Bt, boff[ni] + k0 + 32, half);
    // compute buffer 0
#pragma unroll
    for (int ni = 0; ni < 4; ++ni) {
      c[0][ni] = wmma_bf16(aB[0][0], bB[0][ni], c[0][ni]);
      c[1][ni] = wmma_bf16(aB[0][1], bB[0][ni], c[1][ni]);
    }
    // prefetch buffer 0 @ k0+64
    if (k0 + 64 < K) {
#pragma unroll
      for (int mi = 0; mi < 2; ++mi)
        aB[0][mi] = load_frag_u(A, aoff[mi] + k0 + 64, half);
#pragma unroll
      for (int ni = 0; ni < 4; ++ni)
        bB[0][ni] = load_frag_u(Bt, boff[ni] + k0 + 64, half);
    }
    // compute buffer 1
#pragma unroll
    for (int ni = 0; ni < 4; ++ni) {
      c[0][ni] = wmma_bf16(aB[1][0], bB[1][ni], c[0][ni]);
      c[1][ni] = wmma_bf16(aB[1][1], bB[1][ni], c[1][ni]);
    }
  }

#pragma unroll
  for (int mi = 0; mi < 2; ++mi)
#pragma unroll
    for (int ni = 0; ni < 4; ++ni)
#pragma unroll
      for (int j = 0; j < 8; ++j) {
        int m = m0 + mi * 16 + half * 8 + j;   // C layout: VGPR j, half
        int n = n0 + ni * 16 + (lane & 15);
        size_t idx = (size_t)m * N + n;
        if (MODE == 0) outBF[idx] = f2bf(c[mi][ni][j]);
        else           outF[idx]  = resid[idx] + c[mi][ni][j];
      }
}

// ------- fused gate/up GEMM + SiLU*up epilogue (wave tile 32x32, pipelined) --
__global__ __launch_bounds__(256) void gemm_gateup(
    const __bf16* __restrict__ A, const __bf16* __restrict__ Bg,
    const __bf16* __restrict__ Bu, __bf16* __restrict__ outBF,
    int M, int N, int K) {
  const int lane = threadIdx.x & 31;
  const int half = lane >> 4;
  const int wave = threadIdx.x >> 5;                 // 4 (M) x 2 (N)
  const int m0 = blockIdx.y * 128 + (wave & 3) * 32;
  const int n0 = blockIdx.x * 64 + (wave >> 2) * 32;

  f32x8 cg[2][2], cu[2][2];
#pragma unroll
  for (int mi = 0; mi < 2; ++mi)
#pragma unroll
    for (int ni = 0; ni < 2; ++ni) { cg[mi][ni] = zero8(); cu[mi][ni] = zero8(); }

  unsigned aoff[2], boff[2];
  aoff[0] = (unsigned)(m0 + (lane & 15)) * (unsigned)K;
  aoff[1] = aoff[0] + (unsigned)(16 * K);
#pragma unroll
  for (int ni = 0; ni < 2; ++ni)
    boff[ni] = (unsigned)(n0 + ni * 16 + (lane & 15)) * (unsigned)K;

  bf16x16 aB[2][2], gB[2][2], uB[2][2];
#pragma unroll
  for (int mi = 0; mi < 2; ++mi) aB[0][mi] = load_frag_u(A, aoff[mi], half);
#pragma unroll
  for (int ni = 0; ni < 2; ++ni) {
    gB[0][ni] = load_frag_u(Bg, boff[ni], half);
    uB[0][ni] = load_frag_u(Bu, boff[ni], half);
  }

  for (int k0 = 0; k0 < K; k0 += 64) {
#pragma unroll
    for (int mi = 0; mi < 2; ++mi)
      aB[1][mi] = load_frag_u(A, aoff[mi] + k0 + 32, half);
#pragma unroll
    for (int ni = 0; ni < 2; ++ni) {
      gB[1][ni] = load_frag_u(Bg, boff[ni] + k0 + 32, half);
      uB[1][ni] = load_frag_u(Bu, boff[ni] + k0 + 32, half);
    }
#pragma unroll
    for (int ni = 0; ni < 2; ++ni) {
      cg[0][ni] = wmma_bf16(aB[0][0], gB[0][ni], cg[0][ni]);
      cg[1][ni] = wmma_bf16(aB[0][1], gB[0][ni], cg[1][ni]);
      cu[0][ni] = wmma_bf16(aB[0][0], uB[0][ni], cu[0][ni]);
      cu[1][ni] = wmma_bf16(aB[0][1], uB[0][ni], cu[1][ni]);
    }
    if (k0 + 64 < K) {
#pragma unroll
      for (int mi = 0; mi < 2; ++mi)
        aB[0][mi] = load_frag_u(A, aoff[mi] + k0 + 64, half);
#pragma unroll
      for (int ni = 0; ni < 2; ++ni) {
        gB[0][ni] = load_frag_u(Bg, boff[ni] + k0 + 64, half);
        uB[0][ni] = load_frag_u(Bu, boff[ni] + k0 + 64, half);
      }
    }
#pragma unroll
    for (int ni = 0; ni < 2; ++ni) {
      cg[0][ni] = wmma_bf16(aB[1][0], gB[1][ni], cg[0][ni]);
      cg[1][ni] = wmma_bf16(aB[1][1], gB[1][ni], cg[1][ni]);
      cu[0][ni] = wmma_bf16(aB[1][0], uB[1][ni], cu[0][ni]);
      cu[1][ni] = wmma_bf16(aB[1][1], uB[1][ni], cu[1][ni]);
    }
  }

#pragma unroll
  for (int mi = 0; mi < 2; ++mi)
#pragma unroll
    for (int ni = 0; ni < 2; ++ni)
#pragma unroll
      for (int j = 0; j < 8; ++j) {
        int m = m0 + mi * 16 + half * 8 + j;
        int n = n0 + ni * 16 + (lane & 15);
        float g = cg[mi][ni][j];
        float u = cu[mi][ni][j];
        float silu = g / (1.f + __expf(-g));
        outBF[(size_t)m * N + n] = f2bf(silu * u);
      }
}

// ---------------- RoPE + head split; V stored transposed [b][h][hd][s] -------
__global__ __launch_bounds__(256) void rope_split(
    const __bf16* __restrict__ qraw, const __bf16* __restrict__ kraw,
    const __bf16* __restrict__ vraw, const float* __restrict__ freqs,
    __bf16* __restrict__ Q, __bf16* __restrict__ Kk, __bf16* __restrict__ Vt) {
  int idx = blockIdx.x * 256 + threadIdx.x;
  constexpr int total = CB * CS * CNH * (CHD / 2);   // 2^21
  if (idx >= total) return;
  int i = idx & (CHD / 2 - 1);
  int h = (idx >> 5) & (CNH - 1);
  int s = (idx >> 9) & (CS - 1);
  int b = idx >> 20;

  size_t src = ((size_t)(b * CS + s) * (CNH * CHD)) + h * CHD + 2 * i;
  float cr = freqs[((size_t)s * (CHD / 2) + i) * 2 + 0];
  float ci = freqs[((size_t)s * (CHD / 2) + i) * 2 + 1];
  size_t dst = (((size_t)(b * CNH + h) * CS) + s) * CHD + 2 * i;

  float q0 = bf2f(qraw[src]), q1 = bf2f(qraw[src + 1]);
  Q[dst]     = f2bf(q0 * cr - q1 * ci);
  Q[dst + 1] = f2bf(q0 * ci + q1 * cr);
  float k0 = bf2f(kraw[src]), k1 = bf2f(kraw[src + 1]);
  Kk[dst]     = f2bf(k0 * cr - k1 * ci);
  Kk[dst + 1] = f2bf(k0 * ci + k1 * cr);

  size_t vd = (((size_t)(b * CNH + h) * CHD) + 2 * i) * CS + s;
  Vt[vd]      = vraw[src];
  Vt[vd + CS] = vraw[src + 1];
}

// ---------------- flash attention: one wave per 16-query tile ----------------
__global__ __launch_bounds__(256) void attn_wmma(
    const __bf16* __restrict__ Q, const __bf16* __restrict__ Kk,
    const __bf16* __restrict__ Vt, __bf16* __restrict__ O) {
  __shared__ alignas(16) __bf16 pbuf[8][16 * 32];   // per-wave P tile (bf16)
  const int lane = threadIdx.x & 31;
  const int half = lane >> 4;
  const int wave = threadIdx.x >> 5;
  const int tile = blockIdx.x * 8 + wave;           // 4096 tiles total
  const int qt = tile & (CS / 16 - 1);
  const int bh = tile >> 7;
  const int b = bh >> 4, h = bh & 15;

  const __bf16* qb = Q  + (size_t)bh * CS * CHD;
  const __bf16* kb = Kk + (size_t)bh * CS * CHD;
  const __bf16* vb = Vt + (size_t)bh * CHD * CS;

  const unsigned qoff = (unsigned)(qt * 16 + (lane & 15)) * CHD;
  bf16x16 aq0 = load_frag_u(qb, qoff, half);        // HD 0..31
  bf16x16 aq1 = load_frag_u(qb, qoff + 32, half);   // HD 32..63

  float rmax[8], rsum[8];
  f32x8 o[4];
#pragma unroll
  for (int j = 0; j < 8; ++j) { rmax[j] = -3.0e38f; rsum[j] = 0.f; }
#pragma unroll
  for (int ni = 0; ni < 4; ++ni) o[ni] = zero8();

  const float scale = 0.125f;               // 1/sqrt(64)
  const int kend = qt * 16 + 16;            // causal: keys < kend
  for (int ks = 0; ks < kend; ks += 32) {
    // ---- scores: two 16x16 tiles of Q*K^T ----
    f32x8 s0 = zero8(), s1 = zero8();
    {
      unsigned kr0 = (unsigned)(ks + (lane & 15)) * CHD;
      bf16x16 b0a = load_frag_u(kb, kr0, half);
      bf16x16 b0b = load_frag_u(kb, kr0 + 32, half);
      unsigned kr1 = (unsigned)(ks + 16 + (lane & 15)) * CHD;
      bf16x16 b1a = load_frag_u(kb, kr1, half);
      bf16x16 b1b = load_frag_u(kb, kr1 + 32, half);
      s0 = wmma_bf16(aq0, b0a, s0);
      s0 = wmma_bf16(aq1, b0b, s0);
      s1 = wmma_bf16(aq0, b1a, s1);
      s1 = wmma_bf16(aq1, b1b, s1);
    }
    // ---- online softmax update (rows live in 16-lane halves) ----
#pragma unroll
    for (int j = 0; j < 8; ++j) {
      int qi = qt * 16 + half * 8 + j;
      float v0 = s0[j] * scale, v1 = s1[j] * scale;
      if (ks + (lane & 15) > qi)      v0 = -3.0e38f;
      if (ks + 16 + (lane & 15) > qi) v1 = -3.0e38f;
      float t = fmaxf(v0, v1);
#pragma unroll
      for (int m = 1; m < 16; m <<= 1) t = fmaxf(t, __shfl_xor(t, m, 32));
      float nm = fmaxf(rmax[j], t);
      float alpha = __expf(rmax[j] - nm);
      float p0 = __expf(v0 - nm);
      float p1 = __expf(v1 - nm);
      float ps = p0 + p1;
#pragma unroll
      for (int m = 1; m < 16; m <<= 1) ps += __shfl_xor(ps, m, 32);
      rsum[j] = rsum[j] * alpha + ps;
      rmax[j] = nm;
#pragma unroll
      for (int ni = 0; ni < 4; ++ni) o[ni][j] *= alpha;
      int prow = half * 8 + j;                       // C-layout -> LDS
      pbuf[wave][prow * 32 + (lane & 15)]      = f2bf(p0);
      pbuf[wave][prow * 32 + 16 + (lane & 15)] = f2bf(p1);
    }
    // wave-synchronous LDS round trip (same-wave DS ops are in order)
    asm volatile("s_wait_dscnt 0" ::: "memory");
    bf16x16 ap = load_frag_u(&pbuf[wave][0], (unsigned)(lane & 15) * 32, half);
#pragma unroll
    for (int ni = 0; ni < 4; ++ni) {                 // P(16x32) @ V(32x64)
      bf16x16 bv =
          load_frag_u(vb, (unsigned)(ni * 16 + (lane & 15)) * CS + ks, half);
      o[ni] = wmma_bf16(ap, bv, o[ni]);
    }
    asm volatile("" ::: "memory");
  }

#pragma unroll
  for (int ni = 0; ni < 4; ++ni)
#pragma unroll
    for (int j = 0; j < 8; ++j) {
      int srow = qt * 16 + half * 8 + j;
      int col  = h * CHD + ni * 16 + (lane & 15);
      O[((size_t)(b * CS + srow)) * (CNH * CHD) + col] = f2bf(o[ni][j] / rsum[j]);
    }
}

// ---------------- host orchestration ----------------
extern "C" void kernel_launch(void* const* d_in, const int* in_sizes, int n_in,
                              void* d_out, int out_size, void* d_ws, size_t ws_size,
                              hipStream_t stream) {
  (void)in_sizes; (void)n_in; (void)out_size; (void)ws_size;
  const float* x     = (const float*)d_in[0];
  const float* ropeF = (const float*)d_in[1];
  // d_in[2] = mask (causal, hardcoded)
  const float* attnW = (const float*)d_in[3];
  const float* ffnW  = (const float*)d_in[4];
  const float* wq    = (const float*)d_in[5];
  const float* wk    = (const float*)d_in[6];
  const float* wv    = (const float*)d_in[7];
  const float* wo    = (const float*)d_in[8];
  const float* wg    = (const float*)d_in[9];
  const float* wu    = (const float*)d_in[10];
  const float* wd    = (const float*)d_in[11];
  float* out = (float*)d_out;

  char* p = (char*)d_ws;
  auto allocBF = [&](size_t elems) -> __bf16* {
    __bf16* r = (__bf16*)p;
    p += (elems * sizeof(__bf16) + 255) & ~(size_t)255;
    return r;
  };
  __bf16* wq_t = allocBF((size_t)CD * CD);
  __bf16* wk_t = allocBF((size_t)CD * CD);
  __bf16* wv_t = allocBF((size_t)CD * CD);
  __bf16* wo_t = allocBF((size_t)CD * CD);
  __bf16* wg_t = allocBF((size_t)CD * CIN);
  __bf16* wu_t = allocBF((size_t)CD * CIN);
  __bf16* wd_t = allocBF((size_t)CD * CIN);
  __bf16* h_bf = allocBF((size_t)CM * CD);     // reused as attn_out later
  __bf16* qraw = allocBF((size_t)CM * CD);     // reused as hf later
  __bf16* kraw = allocBF((size_t)CM * CD);
  __bf16* vraw = allocBF((size_t)CM * CD);
  __bf16* Qr   = allocBF((size_t)CM * CD);
  __bf16* Kr   = allocBF((size_t)CM * CD);
  __bf16* Vt   = allocBF((size_t)CM * CD);
  float*  x1   = (float*)p;
  p += ((size_t)CM * CD * sizeof(float) + 255) & ~(size_t)255;
  __bf16* gu   = allocBF((size_t)CM * CIN);

  auto cvt = [&](const float* s, __bf16* d, int R, int C) {
    conv_transpose<<<dim3(C / 32, R / 32), 256, 0, stream>>>(s, d, R, C);
  };
  cvt(wq, wq_t, CD, CD);
  cvt(wk, wk_t, CD, CD);
  cvt(wv, wv_t, CD, CD);
  cvt(wo, wo_t, CD, CD);
  cvt(wg, wg_t, CD, CIN);   // (D,INNER) -> [INNER][D]
  cvt(wu, wu_t, CD, CIN);
  cvt(wd, wd_t, CIN, CD);   // (INNER,D) -> [D][INNER]

  rmsnorm_bf16<<<CM, 256, 0, stream>>>(x, attnW, h_bf);

  dim3 gD(CD / 128, CM / 128);
  gemm_wmma<0><<<gD, 256, 0, stream>>>(h_bf, wq_t, CM, CD, CD, nullptr, nullptr, qraw);
  gemm_wmma<0><<<gD, 256, 0, stream>>>(h_bf, wk_t, CM, CD, CD, nullptr, nullptr, kraw);
  gemm_wmma<0><<<gD, 256, 0, stream>>>(h_bf, wv_t, CM, CD, CD, nullptr, nullptr, vraw);

  int pairs = CB * CS * CNH * (CHD / 2);
  rope_split<<<(pairs + 255) / 256, 256, 0, stream>>>(qraw, kraw, vraw, ropeF,
                                                      Qr, Kr, Vt);

  __bf16* attn_out = h_bf;   // h no longer needed
  attn_wmma<<<(CB * CNH * (CS / 16)) / 8, 256, 0, stream>>>(Qr, Kr, Vt, attn_out);

  gemm_wmma<1><<<gD, 256, 0, stream>>>(attn_out, wo_t, CM, CD, CD, x, x1, nullptr);

  __bf16* hf = qraw;         // qraw no longer needed
  rmsnorm_bf16<<<CM, 256, 0, stream>>>(x1, ffnW, hf);

  gemm_gateup<<<dim3(CIN / 64, CM / 128), 256, 0, stream>>>(hf, wg_t, wu_t, gu,
                                                            CM, CIN, CD);

  gemm_wmma<1><<<gD, 256, 0, stream>>>(gu, wd_t, CM, CD, CIN, x1, out, nullptr);
}
